// LovaszSoftmax_46823733461838
// MI455X (gfx1250) — matrix operand order repaired and checked by reading, not compiled, not run on hardware
//
#include <hip/hip_runtime.h>
#include <stdint.h>

// ---------------- problem constants ----------------
#define C_CLASSES 21
#define B_SZ      8
#define H_SZ      512
#define W_SZ      512
#define HW        (H_SZ * W_SZ)          // 262144
#define P_TOTAL   (B_SZ * HW)            // 2097152 pixels
#define NBINS     4096                   // error-key quantization bins
#define TILE      2048                   // floats per TDM tile (8 KB)
#define SPAN      16384                  // pixels handled per hist block
#define TILES_PER_SPAN (SPAN / TILE)     // 8
#define LOSS_WEIGHT_F 1.0f

typedef unsigned int uint32x4 __attribute__((ext_vector_type(4)));
typedef int          int32x4  __attribute__((ext_vector_type(4)));
typedef int          int32x8  __attribute__((ext_vector_type(8)));
typedef _Float16     v16h     __attribute__((ext_vector_type(16)));
typedef float        v8f      __attribute__((ext_vector_type(8)));

// ---------------------------------------------------------------------------
// TDM: 1-D tile load (n_elems consecutive f32) global -> LDS.
// D# descriptor per CDNA5 ISA ch.8 (§8.3 group0, §8.4 group1).
// UNCONDITIONAL this round: if the builtin is missing we want the compile
// error as signal, not a silent fallback.
// ---------------------------------------------------------------------------
__device__ __forceinline__ void tdm_load_1d(uint32_t lds_byte_addr,
                                            const void* gptr, int n_elems) {
  uint64_t ga = (uint64_t)(uintptr_t)gptr;
  uint32x4 g0;
  g0.x = 1u;                                   // count=1, is_restore=0, gather off
  g0.y = lds_byte_addr;                        // lds_addr [63:32]
  g0.z = (uint32_t)ga;                         // global_addr lo
  g0.w = (uint32_t)((ga >> 32) & 0x1FFFFFFu)   // global_addr [56:32]
       | (2u << 30);                           // type=2 ("image")
  int32x8 g1;
  g1[0] = (2 << 16);                           // data_size=2 (4B); mask=0; no pad/iter
  g1[1] = (int)((n_elems & 0xFFFF) << 16);     // [63:48] tensor_dim0 lo16
  g1[2] = (int)(((n_elems >> 16) & 0xFFFF)     // [79:64] tensor_dim0 hi16
       | (1u << 16));                          // [95:80] tensor_dim1 = 1
  g1[3] = (int)((n_elems & 0xFFFF) << 16);     // [127:112] tile_dim0
  g1[4] = 1;                                   // [143:128] tile_dim1=1, tile_dim2=0
  g1[5] = n_elems;                             // tensor_dim0_stride lo32
  g1[6] = 0;
  g1[7] = 0;
  int32x4 g2 = {0, 0, 0, 0};
  int32x4 g3 = {0, 0, 0, 0};
#if defined(__clang_major__) && (__clang_major__ >= 23)
  int32x8 g4 = {0, 0, 0, 0, 0, 0, 0, 0};
  __builtin_amdgcn_tensor_load_to_lds(g0, g1, g2, g3, g4, 0);
#else
  __builtin_amdgcn_tensor_load_to_lds(g0, g1, g2, g3, 0);
#endif
}

// ---------------------------------------------------------------------------
// Pass A: per-pixel softmax stats. stats[p] = (max_c x, 1/sum_c exp(x-max)).
// ---------------------------------------------------------------------------
__global__ void __launch_bounds__(256)
softmax_stats_kernel(const float* __restrict__ in, float2* __restrict__ stats) {
  int p = blockIdx.x * 256 + threadIdx.x;
  if (p >= P_TOTAL) return;
  int b  = p / HW;
  int hw = p - b * HW;
  const float* base = in + (size_t)b * C_CLASSES * HW + hw;

  float xs[C_CLASSES];
  float m = -3.0e38f;
#pragma unroll
  for (int c = 0; c < C_CLASSES; ++c) {
    if (c + 4 < C_CLASSES)
      __builtin_prefetch(base + (size_t)(c + 4) * HW, 0, 1);  // global_prefetch_b8
    float x = base[(size_t)c * HW];
    xs[c] = x;
    m = fmaxf(m, x);
  }
  float s = 0.0f;
#pragma unroll
  for (int c = 0; c < C_CLASSES; ++c) s += __expf(xs[c] - m);
  stats[p] = make_float2(m, __frcp_rn(s));
}

// ---------------------------------------------------------------------------
// Pass B: per-(class, span) error histogram with LDS privatization.
// Class-plane tiles staged into LDS via the Tensor Data Mover, double-
// buffered on TENSORcnt. Histogram entry packs {fg[hi32] | count[lo32]}.
// ---------------------------------------------------------------------------
__global__ void __launch_bounds__(256)
hist_kernel(const float* __restrict__ in, const float2* __restrict__ stats,
            const int* __restrict__ labels,
            unsigned long long* __restrict__ ghist) {
  __shared__ unsigned long long lhist[NBINS];   // 32 KB
  __shared__ float stage[2][TILE];              // 2 x 8 KB

  const int c    = blockIdx.y;
  const int sp   = blockIdx.x;
  const int spp  = HW / SPAN;                   // spans per plane
  const int b    = sp / spp;
  const int hw0  = (sp - b * spp) * SPAN;
  const int tid  = threadIdx.x;

  for (int i = tid; i < NBINS; i += 256) lhist[i] = 0ULL;

  const float* gsrc  = in + ((size_t)b * C_CLASSES + c) * HW + hw0;
  const int    pbase = b * HW + hw0;

  // Scalar-uniform wave-0 test: forces an s_cbranch so waves 1..7 never
  // reach the TDM instruction (TDM ignores EXEC masking).
  const bool wave0 = (__builtin_amdgcn_readfirstlane((int)threadIdx.x) < 32);
  const uint32_t lds0 = (uint32_t)(uintptr_t)(&stage[0][0]);
  const uint32_t lds1 = (uint32_t)(uintptr_t)(&stage[1][0]);

  if (wave0) tdm_load_1d(lds0, gsrc, TILE);

  for (int t = 0; t < TILES_PER_SPAN; ++t) {
    const int cur = t & 1;
    if (wave0) {
      if (t + 1 < TILES_PER_SPAN) {
        tdm_load_1d(cur ? lds0 : lds1, gsrc + (t + 1) * TILE, TILE);
        __builtin_amdgcn_s_wait_tensorcnt((short)1);   // tile t complete (in-order)
      } else {
        __builtin_amdgcn_s_wait_tensorcnt((short)0);
      }
    }
    __syncthreads();                            // stage[cur] visible to all waves

#pragma unroll
    for (int k = 0; k < TILE / 256; ++k) {
      const int idx = tid + k * 256;
      const float x = stage[cur][idx];
      const int   p = pbase + t * TILE + idx;
      const float2 mr = stats[p];
      const float pc  = __expf(x - mr.x) * mr.y;        // softmax prob of class c
      const int   lab = labels[p];
      const int   fg  = (lab == c) ? 1 : 0;
      const float err = fabsf((float)fg - pc);          // in (0,1)
      int bin = (int)(err * (float)NBINS);
      bin = bin < 0 ? 0 : (bin > NBINS - 1 ? NBINS - 1 : bin);
      const unsigned long long v = 1ULL | ((unsigned long long)fg << 32);
      atomicAdd(&lhist[bin], v);                        // ds_add_u64
    }
    __syncthreads();
  }

  unsigned long long* dst = ghist + (size_t)c * NBINS;
  for (int i = tid; i < NBINS; i += 256) {
    const unsigned long long v = lhist[i];
    if (v) atomicAdd(&dst[i], v);               // global_atomic_add_u64
  }
}

// ---------------------------------------------------------------------------
// Pass C: per-class Lovasz scan via WMMA prefix sums (wave per class).
//
// Inclusive prefix over the 4096 reversed bins, 256 values per round:
// B's 16 columns hold 16 independent 16-bin chunks; A is the 16x16 lower-
// triangular ones matrix (f16-exact) => D[m,n] = within-chunk prefix, f32.
// Counts reach ~2M (> f16), so each sequence is split into base-2048 digit
// planes, scanned separately, recombined as 2048*hi + lo (exact, < 2^24).
// Lovasz telescoping is local: n_{r-1} = n_r - cnt_r, cf_{r-1} = cf_r - fg_r,
// so loss = sum_r e_r * (J(n_r,cf_r) - J(n_r-cnt_r, cf_r-fg_r)).
// ---------------------------------------------------------------------------
__device__ __forceinline__ float jaccard_f(float n, float f, float gts) {
  if (n <= 0.5f) return 0.0f;                 // J_0 := 0 (also guards gts==0)
  const float uni = gts + (n - f);            // > 0 whenever n > 0
  return 1.0f - (gts - f) / uni;
}

__global__ void __launch_bounds__(32)
scan_loss_wmma_kernel(const unsigned long long* __restrict__ ghist,
                      float* __restrict__ classLoss) {
  __shared__ float s_cnt[NBINS];              // reversed-order counts (f32-exact)
  __shared__ float s_fg[NBINS];

  const int c    = blockIdx.x;
  const int lane = threadIdx.x;
  const unsigned long long* h = ghist + (size_t)c * NBINS;

  for (int i = lane; i < NBINS; i += 32) {
    const unsigned long long v = h[NBINS - 1 - i];    // reversed: i=0 <-> max error
    s_cnt[i] = (float)(unsigned int)v;
    s_fg[i]  = (float)(unsigned int)(v >> 32);
  }
  __syncthreads();

  // gts = total foreground count for this class
  float gts = 0.0f;
  for (int i = lane; i < NBINS; i += 32) gts += s_fg[i];
  for (int d = 16; d; d >>= 1) gts += __shfl_down(gts, d);
  gts = __shfl(gts, 0);

  // A: 16x32 f16, A[m,k] = (k <= m && k < 16). Contiguous K order:
  // lanes 0-15: M=lane, K 0..15 in halves h0..h15;
  // lanes16-31: M=lane-16, K 16..31 in halves h0..h15 (all zero).
  const int mrow  = lane & 15;
  const int kbase = (lane < 16) ? 0 : 16;
  v16h A = {};
#pragma unroll
  for (int hh = 0; hh < 16; ++hh) {           // K = kbase + hh
    const int K = kbase + hh;
    A[hh] = (_Float16)((K <= mrow) ? 1.0f : 0.0f);
  }

  const v8f zero = {};
  float carry_c = 0.0f, carry_f = 0.0f, loss = 0.0f;

  for (int q = 0; q < 16; ++q) {
    // B: 32x16 f16. Lanes 0-15: column N=lane = chunk (q*16+lane), rows
    // K=0..15 = that chunk's 16 bin values in halves h0..h15; lanes 16-31
    // (rows K=16..31) stay zero.
    v16h Bcl = {}, Bch = {}, Bfl = {}, Bfh = {};
    if (lane < 16) {
      const int base = q * 256 + lane * 16;
#pragma unroll
      for (int hh = 0; hh < 16; ++hh) {
        const unsigned uc = (unsigned)s_cnt[base + hh];
        const unsigned uf = (unsigned)s_fg[base + hh];
        Bcl[hh] = (_Float16)(float)(uc & 2047u);
        Bch[hh] = (_Float16)(float)(uc >> 11);
        Bfl[hh] = (_Float16)(float)(uf & 2047u);
        Bfh[hh] = (_Float16)(float)(uf >> 11);
      }
    }
    // EXEC all-ones here (whole wave active) as WMMA requires.
    v8f Dcl = __builtin_amdgcn_wmma_f32_16x16x32_f16(false, A, false, Bcl,
                                                     (short)0, zero, false, false);
    v8f Dch = __builtin_amdgcn_wmma_f32_16x16x32_f16(false, A, false, Bch,
                                                     (short)0, zero, false, false);
    v8f Dfl = __builtin_amdgcn_wmma_f32_16x16x32_f16(false, A, false, Bfl,
                                                     (short)0, zero, false, false);
    v8f Dfh = __builtin_amdgcn_wmma_f32_16x16x32_f16(false, A, false, Bfh,
                                                     (short)0, zero, false, false);

    // Recombine digit planes: within-chunk inclusive prefixes, exact f32.
    float sc[8], sf[8];
#pragma unroll
    for (int v = 0; v < 8; ++v) {
      sc[v] = Dch[v] * 2048.0f + Dcl[v];
      sf[v] = Dfh[v] * 2048.0f + Dfl[v];
    }

    // Chunk totals = row M=15 = D-VGPR 7 on lanes 16..31 (N=0..15).
    // Exclusive inter-chunk offsets + round totals via shuffles.
    const int ncol = lane & 15;
    float offc = 0.0f, offf = 0.0f, totc = 0.0f, totf = 0.0f;
#pragma unroll
    for (int j = 0; j < 16; ++j) {
      const float tc = __shfl(sc[7], 16 + j);
      const float tf = __shfl(sf[7], 16 + j);
      if (j < ncol) { offc += tc; offf += tf; }
      totc += tc; totf += tf;
    }

    const int mbase = (lane < 16) ? 0 : 8;    // D layout: lane<16 -> M=v, else M=v+8
#pragma unroll
    for (int v = 0; v < 8; ++v) {
      const float ncum = sc[v] + offc + carry_c;
      const float fcum = sf[v] + offf + carry_f;
      const int   m    = mbase + v;
      const int   r    = q * 256 + ncol * 16 + m;     // reversed linear position
      const float cnt  = s_cnt[r];
      const float fg   = s_fg[r];
      const float J1 = jaccard_f(ncum, fcum, gts);
      const float J0 = jaccard_f(ncum - cnt, fcum - fg, gts);
      const float e  = ((float)(NBINS - 1 - r) + 0.5f) * (1.0f / (float)NBINS);
      loss += e * (J1 - J0);                  // 0 for empty bins automatically
    }
    carry_c += totc;
    carry_f += totf;
  }

  for (int d = 16; d; d >>= 1) loss += __shfl_down(loss, d);
  if (lane == 0) classLoss[c] = loss;
}

// ---------------------------------------------------------------------------
// Pass D: mean over classes.
// ---------------------------------------------------------------------------
__global__ void __launch_bounds__(32)
mean_kernel(const float* __restrict__ classLoss, float* __restrict__ out) {
  const int l = threadIdx.x;
  float v = (l < C_CLASSES) ? classLoss[l] : 0.0f;
  for (int d = 16; d; d >>= 1) v += __shfl_down(v, d);
  if (l == 0) out[0] = LOSS_WEIGHT_F * v * (1.0f / (float)C_CLASSES);
}

// ---------------------------------------------------------------------------
extern "C" void kernel_launch(void* const* d_in, const int* in_sizes, int n_in,
                              void* d_out, int out_size, void* d_ws, size_t ws_size,
                              hipStream_t stream) {
  (void)in_sizes; (void)n_in; (void)out_size; (void)ws_size;
  const float* in     = (const float*)d_in[0];   // [B,C,H,W] f32
  const int*   labels = (const int*)d_in[1];     // [B,H,W]
  float*       out    = (float*)d_out;           // scalar f32

  char* ws = (char*)d_ws;
  float2* stats = (float2*)ws;                                      // 16 MB
  const size_t statsBytes = (size_t)P_TOTAL * sizeof(float2);
  unsigned long long* ghist = (unsigned long long*)(ws + statsBytes);
  const size_t histBytes = (size_t)C_CLASSES * NBINS * sizeof(unsigned long long);
  float* classLoss = (float*)(ws + statsBytes + histBytes);

  hipMemsetAsync(ghist, 0, histBytes, stream);

  softmax_stats_kernel<<<P_TOTAL / 256, 256, 0, stream>>>(in, stats);

  dim3 grid(B_SZ * (HW / SPAN), C_CLASSES);      // (128, 21)
  hist_kernel<<<grid, 256, 0, stream>>>(in, stats, labels, ghist);

  scan_loss_wmma_kernel<<<C_CLASSES, 32, 0, stream>>>(ghist, classLoss);
  mean_kernel<<<1, 32, 0, stream>>>(classLoss, out);
}
